// AnatomicalGCN_13683765805324
// MI455X (gfx1250) — matrix operand order repaired and checked by reading, not compiled, not run on hardware
//
#include <hip/hip_runtime.h>
#include <hip/hip_bf16.h>
#include <math.h>

typedef __attribute__((ext_vector_type(16))) __bf16 v16bf;
typedef __attribute__((ext_vector_type(8)))  float  v8f;
typedef __attribute__((ext_vector_type(4)))  unsigned int v4u;
typedef __attribute__((ext_vector_type(8)))  int    v8i;
typedef __attribute__((ext_vector_type(4)))  int    v4i;

#define GPB   4                 // graphs per block
#define NNODE 12
#define ROWS  (GPB * NNODE)     // 48 rows = 3 WMMA M-tiles
#define HD    256               // hidden dim = 16 N-tiles
#define TD    512
#define EPSLN 1e-5f

__device__ __forceinline__ float gelu_exact(float x) {
  return 0.5f * x * (1.0f + erff(x * 0.70710678118654752440f));
}

// LDS byte offset of a __shared__ object (addrspace(3) offset).
__device__ __forceinline__ unsigned lds_offset(const void* p) {
  return (unsigned)(size_t)(__attribute__((address_space(3))) const char*)p;
}

// TDM: async-load a 2D tile of 32 x 48 f32 (row stride 512 f32) into LDS.
// D# built per CDNA5 ISA ch.8: group0 {count=1, lds_addr, global_addr, type=2},
// group1 {data_size=4B, tensor_dim0=512, tensor_dim1=48, tile_dim0=32,
//          tile_dim1=48, tensor_dim0_stride=512}.
__device__ __forceinline__ void tdm_load_x(const float* gsrc, unsigned lds_addr) {
  const unsigned long long ga = (unsigned long long)(size_t)gsrc;
  v4u g0;
  g0[0] = 1u;                                               // count=1 (valid), no gather
  g0[1] = lds_addr;                                         // lds_addr [63:32]
  g0[2] = (unsigned)ga;                                     // global_addr [95:64]
  g0[3] = (unsigned)((ga >> 32) & 0x1FFFFFFu) | (2u << 30); // global_addr hi | type=2
  v8i g1;
  g1[0] = (int)(2u << 16);                // data_size = 2 -> 4 bytes
  g1[1] = (int)((unsigned)TD << 16);      // tensor_dim0[15:0] = 512
  g1[2] = (int)((unsigned)ROWS << 16);    // tensor_dim0 hi=0 | tensor_dim1 lo=48
  g1[3] = (int)(32u << 16);               // tensor_dim1 hi=0 | tile_dim0=32
  g1[4] = ROWS;                           // tile_dim1=48 | tile_dim2=0
  g1[5] = TD;                             // tensor_dim0_stride = 512
  g1[6] = 0;
  g1[7] = 0;
  const v4i gz4 = {0, 0, 0, 0};
  const v8i gz8 = {0, 0, 0, 0, 0, 0, 0, 0};
  __builtin_amdgcn_tensor_load_to_lds(g0, g1, gz4, gz4, gz8, 0);   // TENSORcnt++
}

// B-fragment of W^T (32K x 16N) from row-major W[rows x ldk]:
// B(k,n) = W[ct*16+n, kt*32+k]; lane L holds col n=L%16, elems e -> k=8*(L/16)+(e&7)+16*(e>>3).
__device__ __forceinline__ v16bf load_b_frag(const float* __restrict__ W, int ldk,
                                             int ct, int kt, int lane) {
  const int row = ct * 16 + (lane & 15);
  const int kb  = kt * 32 + ((lane >> 4) << 3);
  const float4* p = (const float4*)(W + (size_t)row * ldk + kb);
  float4 f0 = p[0], f1 = p[1], f2 = p[4], f3 = p[5];
  v16bf b;
  b[ 0]=(__bf16)f0.x; b[ 1]=(__bf16)f0.y; b[ 2]=(__bf16)f0.z; b[ 3]=(__bf16)f0.w;
  b[ 4]=(__bf16)f1.x; b[ 5]=(__bf16)f1.y; b[ 6]=(__bf16)f1.z; b[ 7]=(__bf16)f1.w;
  b[ 8]=(__bf16)f2.x; b[ 9]=(__bf16)f2.y; b[10]=(__bf16)f2.z; b[11]=(__bf16)f2.w;
  b[12]=(__bf16)f3.x; b[13]=(__bf16)f3.y; b[14]=(__bf16)f3.z; b[15]=(__bf16)f3.w;
  return b;
}

// Cooperatively convert a 48x32 f32 tile into bf16 A-fragment layout:
// lane L of tile rt holds row 16*rt + L%16, group g = L/16,
// element e <-> k: e = (k&7) | ((k&16)>>1), g = (k>>3)&1.
__device__ __forceinline__ void stage_a(const float* __restrict__ src, int ldk,
                                        int kt, __bf16* af, int tid) {
  for (int idx = tid; idx < ROWS * 8; idx += 256) {
    const int row = idx >> 3;
    const int kl  = (idx & 7) * 4;
    const float4 xv = *(const float4*)(src + (size_t)row * ldk + kt * 32 + kl);
    const int g  = (kl >> 3) & 1;
    const int e0 = (kl & 7) | ((kl & 16) >> 1);
    const int L  = g * 16 + (row & 15);
    __bf16* d = af + (((row >> 4) * 32 + L) * 16 + e0);
    d[0] = (__bf16)xv.x; d[1] = (__bf16)xv.y;
    d[2] = (__bf16)xv.z; d[3] = (__bf16)xv.w;
  }
}

// Per-row sum / sum-of-squares over buf[ROWS x HD] (+optional bias) -> ps[row*2+{0,1}]
__device__ __forceinline__ void row_stats(const float* buf, const float* __restrict__ bias,
                                          float* ps, int tid) {
  for (int idx = tid; idx < ROWS * 2; idx += 256) ps[idx] = 0.f;
  __syncthreads();
  for (int idx = tid; idx < ROWS * 8; idx += 256) {
    const int row = idx >> 3, ch = idx & 7;
    const float* p = buf + row * HD + ch * 32;
    float s = 0.f, s2 = 0.f;
#pragma unroll
    for (int c = 0; c < 32; ++c) {
      float v = p[c] + (bias ? bias[ch * 32 + c] : 0.f);
      s += v; s2 += v * v;
    }
    atomicAdd(&ps[row * 2 + 0], s);    // ds_add_f32
    atomicAdd(&ps[row * 2 + 1], s2);
  }
  __syncthreads();
}

__global__ __launch_bounds__(256)
void gat_fused_kernel(const float* __restrict__ x,     const float* __restrict__ Aadj,
                      const float* __restrict__ W_in,  const float* __restrict__ b_in,
                      const float* __restrict__ lnig,  const float* __restrict__ lnib,
                      const float* __restrict__ W0,    const float* __restrict__ a0,
                      const float* __restrict__ W1,    const float* __restrict__ a1,
                      const float* __restrict__ W2,    const float* __restrict__ a2,
                      const float* __restrict__ lng,   const float* __restrict__ lnb,
                      float* __restrict__ out, int Btot) {
  __shared__ float sh_h [ROWS * HD];            // 48 KB: residual activation stream
  __shared__ float sh_wh[ROWS * HD];            // 48 KB: GEMM result / scratch
  __shared__ __align__(16) float sh_xb[2][ROWS * 32];  // 12 KB: TDM double buffer for x
  __shared__ __align__(32) __bf16 sh_af[3 * 32 * 16];  // A fragments for one K-step
  __shared__ float sh_ps [ROWS * 2];            // per-row reductions
  __shared__ float sh_att[GPB * NNODE * NNODE]; // attention logits/weights
  __shared__ float sh_A  [NNODE * NNODE];       // normalized adjacency

  const int tid  = threadIdx.x;
  const int lane = tid & 31;                    // wave32!
  const int w    = tid >> 5;                    // 8 waves
  const int G0   = blockIdx.x * GPB;

  if (tid < NNODE * NNODE) sh_A[tid] = Aadj[tid];

  // ================= GEMM 1: wh = x @ W_in^T  (K=512), TDM-pipelined =========
  v8f acc[6] = {};                              // wave w owns col-tiles {w,w+8} x row-tiles {0,1,2}
  const float* xbase = x + (size_t)G0 * NNODE * TD;
  const unsigned xb_off0 = lds_offset(&sh_xb[0][0]);
  const unsigned xb_off1 = lds_offset(&sh_xb[1][0]);

  if (tid < 32) tdm_load_x(xbase, xb_off0);     // prime the pipeline (wave 0 only)

  for (int kt = 0; kt < TD / 32; ++kt) {
    if (tid < 32) {
      if (kt + 1 < TD / 32) {
        tdm_load_x(xbase + (kt + 1) * 32, (kt + 1) & 1 ? xb_off1 : xb_off0);
        __builtin_amdgcn_s_wait_tensorcnt(1);   // tile kt has landed in LDS
      } else {
        __builtin_amdgcn_s_wait_tensorcnt(0);
      }
    }
    __syncthreads();                            // publish sh_xb[kt&1] to all waves
    stage_a(&sh_xb[kt & 1][0], 32, 0, sh_af, tid);
    __syncthreads();
    __builtin_prefetch(W_in + (size_t)(w * 16 + (lane & 15)) * TD + ((kt + 1) & 15) * 32, 0, 3);
#pragma unroll
    for (int i = 0; i < 2; ++i) {
      const int ct = w + 8 * i;
      const v16bf bf = load_b_frag(W_in, TD, ct, kt, lane);
#pragma unroll
      for (int rt = 0; rt < 3; ++rt) {
        const v16bf af = *(const v16bf*)(sh_af + (rt * 32 + lane) * 16);
        acc[rt * 2 + i] = __builtin_amdgcn_wmma_f32_16x16x32_bf16(
            false, af, false, bf, (short)0, acc[rt * 2 + i], false, false);
      }
    }
    __syncthreads();
  }
#pragma unroll
  for (int i = 0; i < 2; ++i)
#pragma unroll
    for (int rt = 0; rt < 3; ++rt)
#pragma unroll
      for (int v = 0; v < 8; ++v)
        sh_wh[(rt * 16 + v + ((lane >> 4) << 3)) * HD + (w + 8 * i) * 16 + (lane & 15)]
            = acc[rt * 2 + i][v];
  __syncthreads();

  // h = gelu(LN(wh + b_in))
  row_stats(sh_wh, b_in, sh_ps, tid);
  for (int r = 0; r < ROWS; ++r) {
    const float m    = sh_ps[r * 2 + 0] * (1.f / HD);
    const float var  = sh_ps[r * 2 + 1] * (1.f / HD) - m * m;
    const float rstd = rsqrtf(var + EPSLN);
    float v = sh_wh[r * HD + tid] + b_in[tid];
    v = (v - m) * rstd * lnig[tid] + lnib[tid];
    sh_h[r * HD + tid] = gelu_exact(v);
  }
  __syncthreads();

  // ================= 3 GATv2 layers =================
  for (int layer = 0; layer < 3; ++layer) {
    const float* Wl = (layer == 0) ? W0 : (layer == 1) ? W1 : W2;
    const float* al = (layer == 0) ? a0 : (layer == 1) ? a1 : a2;

    // Wh = h @ Wl^T  (K=256)
    v8f acg[6] = {};
    for (int kt = 0; kt < HD / 32; ++kt) {
      stage_a(sh_h, HD, kt, sh_af, tid);
      __syncthreads();
#pragma unroll
      for (int i = 0; i < 2; ++i) {
        const int ct = w + 8 * i;
        const v16bf bf = load_b_frag(Wl, HD, ct, kt, lane);
#pragma unroll
        for (int rt = 0; rt < 3; ++rt) {
          const v16bf af = *(const v16bf*)(sh_af + (rt * 32 + lane) * 16);
          acg[rt * 2 + i] = __builtin_amdgcn_wmma_f32_16x16x32_bf16(
              false, af, false, bf, (short)0, acg[rt * 2 + i], false, false);
        }
      }
      __syncthreads();
    }
#pragma unroll
    for (int i = 0; i < 2; ++i)
#pragma unroll
      for (int rt = 0; rt < 3; ++rt)
#pragma unroll
        for (int v = 0; v < 8; ++v)
          sh_wh[(rt * 16 + v + ((lane >> 4) << 3)) * HD + (w + 8 * i) * 16 + (lane & 15)]
              = acg[rt * 2 + i][v];
    __syncthreads();

    // e_i[row] = Wh[row].a[:256], e_j[row] = Wh[row].a[256:]
    for (int idx = tid; idx < ROWS * 2; idx += 256) sh_ps[idx] = 0.f;
    __syncthreads();
    for (int idx = tid; idx < ROWS * 8; idx += 256) {
      const int row = idx >> 3, ch = idx & 7;
      const float* p  = sh_wh + row * HD + ch * 32;
      const float* ai = al + ch * 32;
      const float* aj = al + HD + ch * 32;
      float si = 0.f, sj = 0.f;
#pragma unroll
      for (int c = 0; c < 32; ++c) { si += p[c] * ai[c]; sj += p[c] * aj[c]; }
      atomicAdd(&sh_ps[row * 2 + 0], si);
      atomicAdd(&sh_ps[row * 2 + 1], sj);
    }
    __syncthreads();

    // logits: leaky_relu(e_i + e_j, 0.2), masked where A==0
    for (int idx = tid; idx < GPB * NNODE * NNODE; idx += 256) {
      const int g = idx / (NNODE * NNODE), r = idx % (NNODE * NNODE);
      const int i = r / NNODE, j = r % NNODE;
      float e = sh_ps[(g * NNODE + i) * 2 + 0] + sh_ps[(g * NNODE + j) * 2 + 1];
      e = (e > 0.f) ? e : 0.2f * e;
      sh_att[idx] = (sh_A[i * NNODE + j] == 0.f) ? -1e30f : e;
    }
    __syncthreads();

    // masked softmax per (graph,row), then * A prior
    if (tid < ROWS) {
      const int g = tid / NNODE, i = tid % NNODE;
      float* er = sh_att + g * NNODE * NNODE + i * NNODE;
      float mx = -1e30f;
#pragma unroll
      for (int j = 0; j < NNODE; ++j) mx = fmaxf(mx, er[j]);
      float pb[NNODE]; float s = 0.f;
#pragma unroll
      for (int j = 0; j < NNODE; ++j) { pb[j] = expf(er[j] - mx); s += pb[j]; }
      const float inv = (s > 0.f) ? 1.f / s : 0.f;
#pragma unroll
      for (int j = 0; j < NNODE; ++j) er[j] = pb[j] * inv * sh_A[i * NNODE + j];
    }
    __syncthreads();

    // h = elu(alpha @ Wh) + h   (thread tid owns column tid for all 48 rows)
    for (int r = 0; r < ROWS; ++r) {
      const int g = r / NNODE, i = r % NNODE;
      const float* arow = sh_att + g * NNODE * NNODE + i * NNODE;
      float s = 0.f;
#pragma unroll
      for (int j = 0; j < NNODE; ++j)
        s += arow[j] * sh_wh[(g * NNODE + j) * HD + tid];
      s = (s > 0.f) ? s : (expf(s) - 1.f);
      sh_h[r * HD + tid] += s;
    }
    __syncthreads();
  }

  // ================= final LN + outputs =================
  row_stats(sh_h, nullptr, sh_ps, tid);
  float* outNode  = out;
  float* outGraph = out + (size_t)Btot * NNODE * HD;
  for (int r = 0; r < ROWS; ++r) {
    const float m    = sh_ps[r * 2 + 0] * (1.f / HD);
    const float var  = sh_ps[r * 2 + 1] * (1.f / HD) - m * m;
    const float rstd = rsqrtf(var + EPSLN);
    const float v = (sh_h[r * HD + tid] - m) * rstd * lng[tid] + lnb[tid];
    sh_wh[r * HD + tid] = v;
    outNode[((size_t)G0 * NNODE + r) * HD + tid] = v;   // coalesced across tid
  }
  __syncthreads();
  for (int g = 0; g < GPB; ++g) {
    float s = 0.f;
#pragma unroll
    for (int i = 0; i < NNODE; ++i) s += sh_wh[(g * NNODE + i) * HD + tid];
    outGraph[(size_t)(G0 + g) * HD + tid] = s * (1.f / NNODE);
  }
}

extern "C" void kernel_launch(void* const* d_in, const int* in_sizes, int n_in,
                              void* d_out, int out_size, void* d_ws, size_t ws_size,
                              hipStream_t stream) {
  (void)n_in; (void)out_size; (void)d_ws; (void)ws_size;
  const float* x    = (const float*)d_in[0];
  const float* Aadj = (const float*)d_in[1];
  const float* W_in = (const float*)d_in[2];
  const float* b_in = (const float*)d_in[3];
  const float* lnig = (const float*)d_in[4];
  const float* lnib = (const float*)d_in[5];
  const float* W0   = (const float*)d_in[6];
  const float* a0   = (const float*)d_in[7];
  const float* W1   = (const float*)d_in[8];
  const float* a1   = (const float*)d_in[9];
  const float* W2   = (const float*)d_in[10];
  const float* a2   = (const float*)d_in[11];
  const float* lng  = (const float*)d_in[12];
  const float* lnb  = (const float*)d_in[13];
  const int Btot = in_sizes[0] / (NNODE * TD);   // 16384
  dim3 grid(Btot / GPB), block(256);
  gat_fused_kernel<<<grid, block, 0, stream>>>(
      x, Aadj, W_in, b_in, lnig, lnib, W0, a0, W1, a1, W2, a2, lng, lnb,
      (float*)d_out, Btot);
}
